// NonLocal2D_7181185319217
// MI455X (gfx1250) — matrix operand order repaired and checked by reading, not compiled, not run on hardware
//
#include <hip/hip_runtime.h>

// ---------------- CDNA5 WMMA types ----------------
typedef __attribute__((ext_vector_type(16))) __bf16 v16bf;
typedef __attribute__((ext_vector_type(2)))  __bf16 v2bf;
typedef __attribute__((ext_vector_type(8)))  float  v8f;
typedef __attribute__((ext_vector_type(4)))  int    v4i;

union Frag { v16bf v; unsigned int u[8]; };
union Acc  { v8f   v; float        f[8]; };

#define NB     8
#define IN_CH  256
#define KD     128
#define S      4096

// ---------------- bf16 conversion helpers ----------------
__device__ __forceinline__ unsigned short f2bf(float a) {
    unsigned int u = __float_as_uint(a);
    return (unsigned short)((u + 0x7FFFu + ((u >> 16) & 1u)) >> 16);
}

#if __has_builtin(__builtin_amdgcn_cvt_pk_bf16_f32)
__device__ __forceinline__ unsigned int pack2bf(float a, float b) {
    v2bf r = __builtin_amdgcn_cvt_pk_bf16_f32(a, b);
    return __builtin_bit_cast(unsigned int, r);
}
#else
__device__ __forceinline__ unsigned int pack2bf(float a, float b) {
    unsigned int ua = __float_as_uint(a);
    unsigned int ub = __float_as_uint(b);
    ua = (ua + 0x7FFFu + ((ua >> 16) & 1u)) >> 16;
    ub = (ub + 0x7FFFu + ((ub >> 16) & 1u)) & 0xFFFF0000u;
    return (ua & 0xFFFFu) | ub;
}
#endif

// gfx1250 async global->LDS copy path (ASYNCcnt-tracked).
// Builtins take typed addrspace(1) src / addrspace(3) dst pointers.
#if __has_builtin(__builtin_amdgcn_global_load_async_to_lds_b128)
#define HAVE_ASYNC_LDS_B128 1
typedef __attribute__((address_space(1))) v4i gv4i;
typedef __attribute__((address_space(3))) v4i lv4i;
#endif
#if __has_builtin(__builtin_amdgcn_global_load_async_to_lds_b32)
#define HAVE_ASYNC_LDS 1
typedef __attribute__((address_space(1))) int gint;
typedef __attribute__((address_space(3))) int lint;
#endif

// A-fragment (16-bit, 16x32) K offset for element v, lane group g.
// Contiguous order matching B fills: lanes 0-15 -> K={0..15},
//             lanes 16-31 -> K={16..31}. Pairs contiguous.
__device__ __forceinline__ int a_koff(int v, int g) {
    return g * 16 + 2 * v;
}

// =====================================================================
// Kernel 0: one-shot f32 -> bf16 conversion of all weight matrices.
// =====================================================================
__global__ __launch_bounds__(256) void wcvt_kernel(
    const float* __restrict__ Wt, const float* __restrict__ Wp,
    const float* __restrict__ Wg, const float* __restrict__ Wo,
    unsigned int* __restrict__ wt2, unsigned int* __restrict__ wp2,
    unsigned int* __restrict__ wg2, unsigned int* __restrict__ wo2)
{
    int i = blockIdx.x * 256 + threadIdx.x;          // over 16384 u32 pairs
    if (i < (KD * IN_CH) / 2) {
        wt2[i] = pack2bf(Wt[2 * i], Wt[2 * i + 1]);
        wp2[i] = pack2bf(Wp[2 * i], Wp[2 * i + 1]);
        wg2[i] = pack2bf(Wg[2 * i], Wg[2 * i + 1]);
        wo2[i] = pack2bf(Wo[2 * i], Wo[2 * i + 1]);
    }
}

// =====================================================================
// Kernel 1: projections.  out[s][k] = sum_c W[k][c] * x[n][c][s] + b[k]
// for W in {Wt (->Q), Wp (->Kt), Wg (->V)}.  Results bf16 [n][S][KD].
// =====================================================================
#define XSTR 264   // bf16 x^T LDS row stride (u16): 528B rows = 16B-aligned

__global__ __launch_bounds__(256) void proj_kernel(
    const float* __restrict__ x,
    const unsigned short* __restrict__ wt2, const float* __restrict__ bt,
    const unsigned short* __restrict__ wp2, const float* __restrict__ bp,
    const unsigned short* __restrict__ wg2, const float* __restrict__ bg,
    unsigned short* __restrict__ qws,
    unsigned short* __restrict__ kws,
    unsigned short* __restrict__ vws)
{
    __shared__ unsigned short xTb[16 * XSTR];        // x^T tile, bf16

    const int n     = blockIdx.y;
    const int sbase = blockIdx.x * 16;
    const int tid   = threadIdx.x;

    // stage x[n][0:256][sbase:sbase+16] transposed + converted into LDS
    for (int idx = tid; idx < 16 * (IN_CH / 2); idx += 256) {
        int so = idx & 15, cp = idx >> 4;            // cp = channel pair
        const float* px = &x[((size_t)(n * IN_CH + 2 * cp)) * S + sbase + so];
        *(unsigned int*)&xTb[so * XSTR + 2 * cp] = pack2bf(px[0], px[S]);
    }
    __syncthreads();

    const int wave = tid >> 5, lane = tid & 31;
    const int ln = lane & 15, g = lane >> 4;
    const int krow = wave * 16 + ln;                 // B-fragment row (8 waves = KD)

    Acc aT, aP, aG;
#pragma unroll
    for (int i = 0; i < 8; ++i) { aT.f[i] = 0.f; aP.f[i] = 0.f; aG.f[i] = 0.f; }

    for (int c0 = 0; c0 < IN_CH; c0 += 32) {
        Frag A;
#pragma unroll
        for (int v = 0; v < 8; ++v)
            A.u[v] = *(const unsigned int*)&xTb[ln * XSTR + c0 + a_koff(v, g)];
        Frag Bt, Bp, Bg;
#pragma unroll
        for (int v = 0; v < 8; ++v) {
            int kk = c0 + g * 16 + 2 * v;
            Bt.u[v] = *(const unsigned int*)&wt2[krow * IN_CH + kk];
            Bp.u[v] = *(const unsigned int*)&wp2[krow * IN_CH + kk];
            Bg.u[v] = *(const unsigned int*)&wg2[krow * IN_CH + kk];
        }
        aT.v = __builtin_amdgcn_wmma_f32_16x16x32_bf16(false, A.v, false, Bt.v,
                                                       (short)0, aT.v, false, false);
        aP.v = __builtin_amdgcn_wmma_f32_16x16x32_bf16(false, A.v, false, Bp.v,
                                                       (short)0, aP.v, false, false);
        aG.v = __builtin_amdgcn_wmma_f32_16x16x32_bf16(false, A.v, false, Bg.v,
                                                       (short)0, aG.v, false, false);
    }

    const float bT = bt[krow], bP = bp[krow], bG = bg[krow];
#pragma unroll
    for (int r = 0; r < 8; ++r) {
        int srow = sbase + r + 8 * g;                // C/D layout: m = r + 8*g
        size_t base = ((size_t)(n * S + srow)) * KD + krow;
        qws[base] = f2bf(aT.f[r] + bT);
        kws[base] = f2bf(aP.f[r] + bP);
        vws[base] = f2bf(aG.f[r] + bG);
    }
}

// =====================================================================
// Kernel 2: fused flash attention + output projection + residual.
// Block: 256 thr (8 waves), 128 query rows.  KV blocks of 64 via LDS.
// All LDS row strides are 16B multiples -> provable alignment -> the
// compiler can merge fragment reads into ds_load_b128.
// =====================================================================
#define KSTR 136   // Kt LDS row stride (u16): 272B = 17*16; rows step 4 banks
#define VSTR 72    // V^T LDS row stride (u16): 144B = 9*16; rows step 36 banks
#define YSTR 136   // per-wave P/y scratch row stride (u16)

__global__ __launch_bounds__(256) void attn_kernel(
    const unsigned short* __restrict__ qws,
    const unsigned short* __restrict__ kws,
    const unsigned short* __restrict__ vws,
    const float* __restrict__ x,
    const unsigned short* __restrict__ wo2, const float* __restrict__ bo,
    float* __restrict__ out)
{
    __shared__ unsigned short Kl[64 * KSTR];        // Kt block [t][k]
    __shared__ unsigned short Vl[KD * VSTR];        // V block transposed [d][t]
    __shared__ unsigned short Yl[8 * 16 * YSTR];    // per-wave P / y scratch

    const int n     = blockIdx.y;
    const int qbase = blockIdx.x * 128;
    const int wave  = threadIdx.x >> 5, lane = threadIdx.x & 31;
    const int ln = lane & 15, g = lane >> 4;
    const int qrow = qbase + wave * 16 + ln;        // A-fragment row
    unsigned short* Pw = &Yl[wave * 16 * YSTR];     // this wave's scratch

    const float SCL = 0.08838834764831845f * 1.44269504088896340f; // (1/sqrt(128))*log2e

    // Q fragments held in registers for the whole KV loop (4 chunks of K=32)
    Frag Qf[4];
#pragma unroll
    for (int c = 0; c < 4; ++c)
#pragma unroll
        for (int v = 0; v < 8; ++v) {
            int k = c * 32 + a_koff(v, g);
            Qf[c].u[v] = *(const unsigned int*)&qws[((size_t)(n * S + qrow)) * KD + k];
        }

    Acc O[8];
    float rm[8], rs[8];
#pragma unroll
    for (int r = 0; r < 8; ++r) {
        rm[r] = -3.0e38f; rs[r] = 0.f;
#pragma unroll
        for (int jo = 0; jo < 8; ++jo) O[jo].f[r] = 0.f;
    }

    for (int kv = 0; kv < S; kv += 64) {
        __syncthreads();
        // ---- stage Kt block: async global->LDS (ASYNCcnt) ----
#if HAVE_ASYNC_LDS_B128
        for (int idx = threadIdx.x; idx < 64 * 16; idx += 256) {
            int t = idx >> 4, ck = (idx & 15) * 8;   // 16B per lane
            __builtin_amdgcn_global_load_async_to_lds_b128(
                (gv4i*)&kws[((size_t)(n * S + kv + t)) * KD + ck],
                (lv4i*)&Kl[t * KSTR + ck], 0, 0);
        }
#elif defined(HAVE_ASYNC_LDS)
        for (int idx = threadIdx.x; idx < 64 * 64; idx += 256) {
            int t = idx >> 6, kk = (idx & 63) * 2;
            __builtin_amdgcn_global_load_async_to_lds_b32(
                (gint*)&kws[((size_t)(n * S + kv + t)) * KD + kk],
                (lint*)&Kl[t * KSTR + kk], 0, 0);
        }
#else
        for (int idx = threadIdx.x; idx < 64 * 64; idx += 256) {
            int t = idx >> 6, kk = (idx & 63) * 2;
            *(unsigned int*)&Kl[t * KSTR + kk] =
                *(const unsigned int*)&kws[((size_t)(n * S + kv + t)) * KD + kk];
        }
#endif
        // ---- stage V block transposed: Vl[d][t] = V[kv+t][d] ----
        for (int idx = threadIdx.x; idx < 64 * (KD / 2); idx += 256) {
            int dp = idx & 63, t = idx >> 6;         // dp = channel pair
            unsigned int w = *(const unsigned int*)
                &vws[((size_t)(n * S + kv + t)) * KD + 2 * dp];
            Vl[(2 * dp)     * VSTR + t] = (unsigned short)(w & 0xFFFFu);
            Vl[(2 * dp + 1) * VSTR + t] = (unsigned short)(w >> 16);
        }
#if HAVE_ASYNC_LDS_B128 || defined(HAVE_ASYNC_LDS)
#if __has_builtin(__builtin_amdgcn_s_wait_asynccnt)
        __builtin_amdgcn_s_wait_asynccnt(0);
#else
        asm volatile("s_wait_asynccnt 0x0" ::: "memory");
#endif
#endif
        __syncthreads();

        // ---- scores: S[16 x 64] per wave = Q(16x128) * Kt^T ----
        Acc Sv[4];
#pragma unroll
        for (int j = 0; j < 4; ++j) {
            Acc a;
#pragma unroll
            for (int i = 0; i < 8; ++i) a.f[i] = 0.f;
#pragma unroll
            for (int c = 0; c < 4; ++c) {
                Frag B;
#pragma unroll
                for (int v = 0; v < 8; ++v) {
                    int kk = c * 32 + g * 16 + 2 * v;
                    B.u[v] = *(const unsigned int*)&Kl[(j * 16 + ln) * KSTR + kk];
                }
                a.v = __builtin_amdgcn_wmma_f32_16x16x32_bf16(false, Qf[c].v, false, B.v,
                                                              (short)0, a.v, false, false);
            }
            Sv[j] = a;
        }

        // ---- online softmax (log2 domain, scale folded in) ----
        float nm[8];
#pragma unroll
        for (int r = 0; r < 8; ++r) {
            float m = rm[r];
#pragma unroll
            for (int j = 0; j < 4; ++j) m = fmaxf(m, Sv[j].f[r] * SCL);
            nm[r] = m;
        }
#pragma unroll
        for (int off = 1; off < 16; off <<= 1)
#pragma unroll
            for (int r = 0; r < 8; ++r)
                nm[r] = fmaxf(nm[r], __shfl_xor(nm[r], off, 32));
#pragma unroll
        for (int r = 0; r < 8; ++r) {
            float corr = __builtin_amdgcn_exp2f(rm[r] - nm[r]);
            rm[r] = nm[r];
            rs[r] *= corr;
#pragma unroll
            for (int jo = 0; jo < 8; ++jo) O[jo].f[r] *= corr;
        }

        // P = exp2(S*SCL - m), store to per-wave LDS (row=m, col=t)
        float ps[8];
#pragma unroll
        for (int r = 0; r < 8; ++r) ps[r] = 0.f;
#pragma unroll
        for (int j = 0; j < 4; ++j)
#pragma unroll
            for (int r = 0; r < 8; ++r) {
                float p = __builtin_amdgcn_exp2f(Sv[j].f[r] * SCL - rm[r]);
                ps[r] += p;
                Pw[(r + 8 * g) * YSTR + j * 16 + ln] = f2bf(p);
            }
#pragma unroll
        for (int off = 1; off < 16; off <<= 1)
#pragma unroll
            for (int r = 0; r < 8; ++r)
                ps[r] += __shfl_xor(ps[r], off, 32);
#pragma unroll
        for (int r = 0; r < 8; ++r) rs[r] += ps[r];

        // cross-lane LDS RAW: force DS completion + compiler ordering
        asm volatile("s_wait_dscnt 0x0" ::: "memory");

        // ---- O += P(16x64) * V(64x128) ----
#pragma unroll
        for (int jo = 0; jo < 8; ++jo) {
#pragma unroll
            for (int c2 = 0; c2 < 2; ++c2) {
                Frag A, B;
#pragma unroll
                for (int v = 0; v < 8; ++v)
                    A.u[v] = *(const unsigned int*)&Pw[ln * YSTR + c2 * 32 + a_koff(v, g)];
#pragma unroll
                for (int v = 0; v < 8; ++v) {
                    int tt = c2 * 32 + g * 16 + 2 * v;
                    B.u[v] = *(const unsigned int*)&Vl[(jo * 16 + ln) * VSTR + tt];
                }
                O[jo].v = __builtin_amdgcn_wmma_f32_16x16x32_bf16(false, A.v, false, B.v,
                                                                  (short)0, O[jo].v, false, false);
            }
        }
    }

    // ---- normalize, stage y (bf16) into per-wave scratch ----
#pragma unroll
    for (int r = 0; r < 8; ++r) rs[r] = 1.0f / rs[r];
#pragma unroll
    for (int jo = 0; jo < 8; ++jo)
#pragma unroll
        for (int r = 0; r < 8; ++r)
            Pw[(r + 8 * g) * YSTR + jo * 16 + ln] = f2bf(O[jo].f[r] * rs[r]);

    asm volatile("s_wait_dscnt 0x0" ::: "memory");

    // ---- out[c][s] = x + y(16x128) * Wo^T(128x256) + bo ----
    for (int ct = 0; ct < 16; ++ct) {
        Acc a2;
#pragma unroll
        for (int i = 0; i < 8; ++i) a2.f[i] = 0.f;
#pragma unroll
        for (int c2 = 0; c2 < 4; ++c2) {
            Frag A, B;
#pragma unroll
            for (int v = 0; v < 8; ++v)
                A.u[v] = *(const unsigned int*)&Pw[ln * YSTR + c2 * 32 + a_koff(v, g)];
#pragma unroll
            for (int v = 0; v < 8; ++v) {
                int kk = c2 * 32 + g * 16 + 2 * v;
                B.u[v] = *(const unsigned int*)&wo2[(ct * 16 + ln) * KD + kk];
            }
            a2.v = __builtin_amdgcn_wmma_f32_16x16x32_bf16(false, A.v, false, B.v,
                                                           (short)0, a2.v, false, false);
        }
        int c = ct * 16 + ln;
        float bias = bo[c];
#pragma unroll
        for (int r = 0; r < 8; ++r) {
            int srow = qbase + wave * 16 + r + 8 * g;   // 8 consecutive s per lane
            size_t oi = ((size_t)(n * IN_CH + c)) * S + srow;
            out[oi] = a2.f[r] + x[oi] + bias;
        }
    }
}

extern "C" void kernel_launch(void* const* d_in, const int* in_sizes, int n_in,
                              void* d_out, int out_size, void* d_ws, size_t ws_size,
                              hipStream_t stream) {
    const float* x  = (const float*)d_in[0];
    const float* Wg = (const float*)d_in[1];
    const float* bg = (const float*)d_in[2];
    const float* Wt = (const float*)d_in[3];
    const float* bt = (const float*)d_in[4];
    const float* Wp = (const float*)d_in[5];
    const float* bp = (const float*)d_in[6];
    const float* Wo = (const float*)d_in[7];
    const float* bo = (const float*)d_in[8];
    float* out = (float*)d_out;

    const size_t SZ = (size_t)NB * S * KD;            // 4 Mi elements per tensor
    const size_t WSZ = (size_t)KD * IN_CH;            // 32768 elements per weight
    unsigned short* qws = (unsigned short*)d_ws;      // theta -> Q   [n][S][K] bf16
    unsigned short* kws = qws + SZ;                   // phi   -> Kt  [n][S][K] bf16
    unsigned short* vws = kws + SZ;                   // g     -> V   [n][S][K] bf16
    unsigned short* wt2 = vws + SZ;                   // bf16 weights
    unsigned short* wp2 = wt2 + WSZ;
    unsigned short* wg2 = wp2 + WSZ;
    unsigned short* wo2 = wg2 + WSZ;

    wcvt_kernel<<<dim3((WSZ / 2 + 255) / 256), 256, 0, stream>>>(
        Wt, Wp, Wg, Wo,
        (unsigned int*)wt2, (unsigned int*)wp2,
        (unsigned int*)wg2, (unsigned int*)wo2);
    proj_kernel<<<dim3(S / 16, NB), 256, 0, stream>>>(
        x, wt2, bt, wp2, bp, wg2, bg, qws, kws, vws);
    attn_kernel<<<dim3(S / 128, NB), 256, 0, stream>>>(
        qws, kws, vws, x, wo2, bo, out);
}
